// GraphGCN_48911087567500
// MI455X (gfx1250) — matrix operand order repaired and checked by reading, not compiled, not run on hardware
//
#include <hip/hip_runtime.h>
#include <cstdint>
#include <cstddef>

// ---------------- problem constants ----------------
#define N_TWEETS 4096
#define N_USERS  2048
#define NN       6144            // N_NODES
#define NE       98304           // E edges
#define SEQ      32
#define EMBED    300
#define HID      100
#define G3       300             // 3*HID
#define H1C      64
#define H2C      100
#define UFEAT    9
#define BS       2048
#define MROWS    (SEQ * N_TWEETS)    // 131072
#define KP_EMB   320                 // EMBED padded to 32
#define KP_HID   128                 // HID padded to 32
#define NP_G3    320                 // 3*HID padded to 32
#define NP_H1    64
#define NP_H2    128
// pos_weight is analytic: A_ground.sum() == E + N always
#define POSW     ((37748736.0f - 104448.0f) / 104448.0f)

typedef __bf16 bf16;
typedef __attribute__((ext_vector_type(16))) __bf16 v16bf;
typedef __attribute__((ext_vector_type(8)))  __bf16 v8bf;
typedef __attribute__((ext_vector_type(8)))  float   v8f;

__device__ __forceinline__ float sigmoidf_(float x) { return 1.0f / (1.0f + __expf(-x)); }

// CDNA5 async global->LDS 16B copy (ASYNCcnt-tracked DMA, no VGPR staging)
__device__ __forceinline__ void async_cp16(const bf16* g, bf16* l)
{
    asm volatile("global_load_async_to_lds_b128 %0, %1, off"
                 :: "v"((uint32_t)(uintptr_t)l), "v"(g)
                 : "memory");
}
__device__ __forceinline__ void wait_async0()
{
    asm volatile("s_wait_asynccnt 0x0" ::: "memory");
}

// ---------------------------------------------------------------------------
// bf16 WMMA GEMM: C[M, Nact](f32) = A[M, lda](bf16) * B[Npad, ldb]^T (+bias)
// Block = 256 thr = 8 waves; tile 64(M) x 64(N); each wave: 1 A-frag x 2 B-frags
// per 32-K chunk (2 WMMAs). LDS double-buffered, filled via async DMA.
// ---------------------------------------------------------------------------
__global__ void k_gemm_bf16(const bf16* __restrict__ A, int lda,
                            const bf16* __restrict__ B, int ldb,
                            const float* __restrict__ bias,
                            float* __restrict__ C, int Nact, int Kp)
{
    __shared__ bf16 As[2][64 * 40];   // 80B row stride (bank spread, 16B aligned)
    __shared__ bf16 Bs[2][64 * 40];
    const int t    = threadIdx.x;
    const int lane = t & 31;
    const int wv   = t >> 5;
    const int wm   = wv >> 1;          // 0..3  -> 16-row M sub-tile
    const int wn   = wv & 1;           // 0..1  -> 32-col N half
    const int hlf  = lane >> 4;
    const int l15  = lane & 15;
    const int m0   = blockIdx.x * 64;
    const int n0   = blockIdx.y * 64;
    const int row  = t >> 2, cg = (t & 3) * 8;   // 256 thr cover 64x32 bf16 tile

    const bf16* Agp = &A[(long)(m0 + row) * lda + cg];
    const bf16* Bgp = &B[(long)(n0 + row) * ldb + cg];

    // prefetch chunk 0 into buffer 0
    async_cp16(Agp, &As[0][row * 40 + cg]);
    async_cp16(Bgp, &Bs[0][row * 40 + cg]);
    wait_async0();
    __syncthreads();

    v8f acc0 = {}, acc1 = {};
    const int nK = Kp >> 5;
    for (int k = 0; k < nK; ++k) {
        const int cur = k & 1, nxt = cur ^ 1;
        if (k + 1 < nK) {                       // async-issue next chunk
            async_cp16(Agp + (k + 1) * 32, &As[nxt][row * 40 + cg]);
            async_cp16(Bgp + (k + 1) * 32, &Bs[nxt][row * 40 + cg]);
        }
        // A fragment: lane row l15, K runs [hlf*8,+8) and [16+hlf*8,+8)
        const bf16* ap = &As[cur][(wm * 16 + l15) * 40];
        v8bf alo = *(const v8bf*)(ap + hlf * 8);
        v8bf ahi = *(const v8bf*)(ap + 16 + hlf * 8);
        v16bf a;
        #pragma unroll
        for (int i = 0; i < 8; ++i) { a[i] = alo[i]; a[8 + i] = ahi[i]; }
        // Two B fragments: lane col l15, K runs [hlf*8,+8) and [16+hlf*8,+8)
        const bf16* bp0 = &Bs[cur][(wn * 32 + l15) * 40 + hlf * 8];
        const bf16* bp1 = &Bs[cur][(wn * 32 + 16 + l15) * 40 + hlf * 8];
        v8bf b0l = *(const v8bf*)bp0, b0h = *(const v8bf*)(bp0 + 16);
        v8bf b1l = *(const v8bf*)bp1, b1h = *(const v8bf*)(bp1 + 16);
        v16bf b0, b1;
        #pragma unroll
        for (int i = 0; i < 8; ++i) {
            b0[i] = b0l[i]; b0[8 + i] = b0h[i];
            b1[i] = b1l[i]; b1[8 + i] = b1h[i];
        }
        acc0 = __builtin_amdgcn_wmma_f32_16x16x32_bf16(false, a, false, b0,
                                                       (short)0, acc0, false, false);
        acc1 = __builtin_amdgcn_wmma_f32_16x16x32_bf16(false, a, false, b1,
                                                       (short)0, acc1, false, false);
        wait_async0();          // next buffer resident before the swap barrier
        __syncthreads();
    }
    #pragma unroll
    for (int sub = 0; sub < 2; ++sub) {
        const v8f acc = sub ? acc1 : acc0;
        const int gn = n0 + wn * 32 + sub * 16 + l15;
        if (gn < Nact) {
            float bv = bias ? bias[gn] : 0.0f;
            #pragma unroll
            for (int i = 0; i < 8; ++i) {
                int gm = m0 + wm * 16 + hlf * 8 + i;
                C[(long)gm * Nact + gn] = acc[i] + bv;
            }
        }
    }
}

// ---------------------------------------------------------------------------
// Fused Z*Z^T -> sigmoid -> weighted BCE + A_pred sum (same WMMA core, 64x64).
// accum[0] += sum(sigmoid(logit));  accum[1] += sum(weight*bce)
// ---------------------------------------------------------------------------
__global__ void k_adj_loss(const bf16* __restrict__ Zb,
                           const float* __restrict__ Ag,
                           float* __restrict__ accum)
{
    __shared__ bf16 As[2][64 * 40];
    __shared__ bf16 Bs[2][64 * 40];
    __shared__ float redA[256];
    __shared__ float redB[256];
    const int t = threadIdx.x, lane = t & 31, wv = t >> 5;
    const int wm = wv >> 1, wn = wv & 1, hlf = lane >> 4, l15 = lane & 15;
    const int m0 = blockIdx.x * 64, n0 = blockIdx.y * 64;
    const int row = t >> 2, cg = (t & 3) * 8;

    const bf16* Agp = &Zb[(long)(m0 + row) * KP_HID + cg];
    const bf16* Bgp = &Zb[(long)(n0 + row) * KP_HID + cg];
    async_cp16(Agp, &As[0][row * 40 + cg]);
    async_cp16(Bgp, &Bs[0][row * 40 + cg]);
    wait_async0();
    __syncthreads();

    v8f acc0 = {}, acc1 = {};
    const int nK = KP_HID >> 5;
    for (int k = 0; k < nK; ++k) {
        const int cur = k & 1, nxt = cur ^ 1;
        if (k + 1 < nK) {
            async_cp16(Agp + (k + 1) * 32, &As[nxt][row * 40 + cg]);
            async_cp16(Bgp + (k + 1) * 32, &Bs[nxt][row * 40 + cg]);
        }
        const bf16* ap = &As[cur][(wm * 16 + l15) * 40];
        v8bf alo = *(const v8bf*)(ap + hlf * 8);
        v8bf ahi = *(const v8bf*)(ap + 16 + hlf * 8);
        v16bf a;
        #pragma unroll
        for (int i = 0; i < 8; ++i) { a[i] = alo[i]; a[8 + i] = ahi[i]; }
        const bf16* bp0 = &Bs[cur][(wn * 32 + l15) * 40 + hlf * 8];
        const bf16* bp1 = &Bs[cur][(wn * 32 + 16 + l15) * 40 + hlf * 8];
        v8bf b0l = *(const v8bf*)bp0, b0h = *(const v8bf*)(bp0 + 16);
        v8bf b1l = *(const v8bf*)bp1, b1h = *(const v8bf*)(bp1 + 16);
        v16bf b0, b1;
        #pragma unroll
        for (int i = 0; i < 8; ++i) {
            b0[i] = b0l[i]; b0[8 + i] = b0h[i];
            b1[i] = b1l[i]; b1[8 + i] = b1h[i];
        }
        acc0 = __builtin_amdgcn_wmma_f32_16x16x32_bf16(false, a, false, b0,
                                                       (short)0, acc0, false, false);
        acc1 = __builtin_amdgcn_wmma_f32_16x16x32_bf16(false, a, false, b1,
                                                       (short)0, acc1, false, false);
        wait_async0();
        __syncthreads();
    }

    float s_pred = 0.0f, s_wbce = 0.0f;
    #pragma unroll
    for (int sub = 0; sub < 2; ++sub) {
        const v8f acc = sub ? acc1 : acc0;
        const int gn = n0 + wn * 32 + sub * 16 + l15;
        #pragma unroll
        for (int i = 0; i < 8; ++i) {
            int gm = m0 + wm * 16 + hlf * 8 + i;
            float pred = sigmoidf_(acc[i]);
            float p = fminf(fmaxf(pred, 1e-7f), 1.0f - 1e-7f);
            float g = Ag[(long)gm * NN + gn] + ((gm == gn) ? 1.0f : 0.0f);
            float w = (g == 1.0f) ? POSW : 1.0f;
            float bce = -(g * __logf(p) + (1.0f - g) * log1pf(-p));
            s_pred += pred;
            s_wbce += w * bce;
        }
    }
    redA[t] = s_pred; redB[t] = s_wbce;
    __syncthreads();
    for (int off = 128; off > 0; off >>= 1) {
        if (t < off) { redA[t] += redA[t + off]; redB[t] += redB[t + off]; }
        __syncthreads();
    }
    if (t == 0) { atomicAdd(&accum[0], redA[0]); atomicAdd(&accum[1], redB[0]); }
}

// ---------------------------------------------------------------------------
// Small helper kernels
// ---------------------------------------------------------------------------
__global__ void k_pack(const float* __restrict__ W, int rows, int cols,
                       int Np, int Kp, int trans, bf16* __restrict__ out)
{
    int idx = blockIdx.x * 256 + threadIdx.x;
    if (idx >= Np * Kp) return;
    int n = idx / Kp, k = idx % Kp;
    float v = 0.0f;
    if (!trans) { if (n < rows && k < cols) v = W[n * cols + k]; }
    else        { if (k < rows && n < cols) v = W[k * cols + n]; }
    out[idx] = (bf16)v;
}

__global__ void k_cvt_pad(const float* __restrict__ in, bf16* __restrict__ out,
                          int M, int K, int Kp)
{
    int idx = blockIdx.x * 256 + threadIdx.x;
    if (idx >= M * Kp) return;
    int i = idx / Kp, j = idx % Kp;
    out[idx] = (j < K) ? (bf16)in[i * K + j] : (bf16)0.0f;
}

__global__ void k_embed_gather(const int* __restrict__ gnf,
                               const float* __restrict__ emb,
                               bf16* __restrict__ X)
{
    int r = blockIdx.x;
    int s = r >> 12;
    int tt = r & 4095;
    int tok = gnf[tt * SEQ + s];
    const float* src = emb + (long)tok * EMBED;
    bf16* dst = X + (long)r * KP_EMB;
    for (int c = threadIdx.x; c < KP_EMB; c += 64)
        dst[c] = (c < EMBED) ? (bf16)src[c] : (bf16)0.0f;
}

__global__ void k_user_mlp(const float* __restrict__ uf,
                           const float* __restrict__ w1, const float* __restrict__ b1,
                           const float* __restrict__ w2, const float* __restrict__ b2,
                           float* __restrict__ ue)
{
    __shared__ float hid[100];
    int r = blockIdx.x, t = threadIdx.x;
    if (t < 100) {
        float s = b1[t];
        #pragma unroll
        for (int k = 0; k < UFEAT; ++k) s += uf[r * UFEAT + k] * w1[k * 100 + t];
        hid[t] = fmaxf(s, 0.0f);
    }
    __syncthreads();
    if (t < 100) {
        float s = b2[t];
        for (int k = 0; k < 100; ++k) s += hid[k] * w2[k * 100 + t];
        ue[r * 100 + t] = s;
    }
}

__global__ void k_gates(const float* __restrict__ gi, const float* __restrict__ gh,
                        const float* __restrict__ hprev, float* __restrict__ hout,
                        bf16* __restrict__ hbf)
{
    int idx = blockIdx.x * 256 + threadIdx.x;
    if (idx >= N_TWEETS * HID) return;
    int tt = idx / HID, j = idx % HID;
    const float* gir = gi + (long)tt * G3;
    const float* ghr = gh + (long)tt * G3;
    float r = sigmoidf_(gir[j] + ghr[j]);
    float z = sigmoidf_(gir[HID + j] + ghr[HID + j]);
    float n = tanhf(gir[2 * HID + j] + r * ghr[2 * HID + j]);
    float h = hprev[idx];
    float hn = (1.0f - z) * n + z * h;
    hout[idx] = hn;
    hbf[(long)tt * KP_HID + j] = (bf16)hn;
}

__global__ void k_build_x(const float* __restrict__ hn, const float* __restrict__ ue,
                          bf16* __restrict__ xinb)
{
    int idx = blockIdx.x * 256 + threadIdx.x;
    if (idx >= NN * KP_HID) return;
    int i = idx >> 7, j = idx & 127;
    float v = 0.0f;
    if (j < HID) {
        if (i < BS)                 v = hn[i * HID + j];
        else if (i < BS + N_USERS)  v = ue[(i - BS) * HID + j];
        else                        v = hn[(i - N_USERS) * HID + j];
    }
    xinb[idx] = (bf16)v;
}

__global__ void k_deg(const int* __restrict__ ei, float* __restrict__ deg)
{
    int e = blockIdx.x * 256 + threadIdx.x;
    if (e >= NE) return;
    atomicAdd(&deg[ei[NE + e]], 1.0f);
}

__global__ void k_ground(const int* __restrict__ ei, float* __restrict__ Ag)
{
    int e = blockIdx.x * 256 + threadIdx.x;
    if (e >= NE) return;
    atomicAdd(&Ag[(long)ei[e] * NN + ei[NE + e]], 1.0f);
}

__global__ void k_dis(const float* __restrict__ deg, float* __restrict__ dis)
{
    int i = blockIdx.x * 256 + threadIdx.x;
    if (i >= NN) return;
    dis[i] = rsqrtf(deg[i] + 1.0f);
}

__global__ void k_scatter(const float* __restrict__ xw, float* __restrict__ agg,
                          const int* __restrict__ ei, const float* __restrict__ dis, int F)
{
    long idx = (long)blockIdx.x * 256 + threadIdx.x;
    if (idx >= (long)NE * F) return;
    int e = (int)(idx / F), f = (int)(idx % F);
    int s = ei[e], d = ei[NE + e];
    atomicAdd(&agg[(long)d * F + f], xw[(long)s * F + f] * dis[s] * dis[d]);
}

__global__ void k_final_elu(const float* __restrict__ xw, const float* __restrict__ agg,
                            const float* __restrict__ dis, const float* __restrict__ b,
                            bf16* __restrict__ outb)
{
    int idx = blockIdx.x * 256 + threadIdx.x;
    if (idx >= NN * H1C) return;
    int i = idx / H1C, f = idx % H1C;
    float v = agg[idx] + xw[idx] * dis[i] * dis[i] + b[f];
    outb[idx] = (bf16)(v > 0.0f ? v : expm1f(v));
}

__global__ void k_final_relu(const float* __restrict__ xw, const float* __restrict__ agg,
                             const float* __restrict__ dis, const float* __restrict__ b,
                             float* __restrict__ out)
{
    int idx = blockIdx.x * 256 + threadIdx.x;
    if (idx >= NN * H2C) return;
    int i = idx / H2C, f = idx % H2C;
    float v = agg[idx] + xw[idx] * dis[i] * dis[i] + b[f];
    out[idx] = fmaxf(v, 0.0f);
}

__global__ void k_z(const float* __restrict__ mu, const float* __restrict__ lv,
                    const float* __restrict__ eps, bf16* __restrict__ Zb,
                    float* __restrict__ out, float* __restrict__ accum)
{
    __shared__ float red[256];
    int t = threadIdx.x;
    int idx = blockIdx.x * 256 + t;
    float klp = 0.0f;
    if (idx < NN * KP_HID) {
        int i = idx >> 7, j = idx & 127;
        if (j < H2C) {
            float m = mu[i * H2C + j], l = lv[i * H2C + j];
            float z = m + __expf(0.5f * l) * eps[i * H2C + j];
            Zb[idx] = (bf16)z;
            if (i < BS) out[i * H2C + j] = z;
            klp = -0.5f * (1.0f + l - m * m - __expf(l));
        } else {
            Zb[idx] = (bf16)0.0f;
        }
    }
    red[t] = klp;
    __syncthreads();
    for (int off = 128; off > 0; off >>= 1) {
        if (t < off) red[t] += red[t + off];
        __syncthreads();
    }
    if (t == 0) atomicAdd(&accum[2], red[0]);
}

__global__ void k_final(const float* __restrict__ accum, float* __restrict__ out)
{
    if (threadIdx.x == 0) {
        const float n2 = 37748736.0f;      // 6144^2
        out[BS * H2C]     = accum[2] / (float)NN;                 // kl_loss
        float norm = n2 / ((n2 - accum[0]) * 2.0f);
        out[BS * H2C + 1] = norm * (accum[1] / n2);               // rec_loss
    }
}

// ---------------------------------------------------------------------------
extern "C" void kernel_launch(void* const* d_in, const int* in_sizes, int n_in,
                              void* d_out, int out_size, void* d_ws, size_t ws_size,
                              hipStream_t stream)
{
    (void)in_sizes; (void)n_in; (void)out_size; (void)ws_size;
    const float* uf    = (const float*)d_in[0];
    const int*   gnf   = (const int*)d_in[1];
    const int*   ei    = (const int*)d_in[2];
    const float* h0    = (const float*)d_in[4];
    const float* eps   = (const float*)d_in[5];
    const float* emb   = (const float*)d_in[6];
    const float* w1    = (const float*)d_in[7];
    const float* b1    = (const float*)d_in[8];
    const float* w2    = (const float*)d_in[9];
    const float* b2    = (const float*)d_in[10];
    const float* wih0  = (const float*)d_in[11];
    const float* whh0  = (const float*)d_in[12];
    const float* bih0  = (const float*)d_in[13];
    const float* bhh0  = (const float*)d_in[14];
    const float* wih1  = (const float*)d_in[15];
    const float* whh1  = (const float*)d_in[16];
    const float* bih1  = (const float*)d_in[17];
    const float* bhh1  = (const float*)d_in[18];
    const float* c1w   = (const float*)d_in[19];
    const float* c1b   = (const float*)d_in[20];
    const float* cmw   = (const float*)d_in[21];
    const float* cmb   = (const float*)d_in[22];
    const float* clw   = (const float*)d_in[23];
    const float* clb   = (const float*)d_in[24];
    float* out = (float*)d_out;
    char*  ws  = (char*)d_ws;

    // -------- workspace layout --------
    size_t off = 0;
    auto alloc = [&](size_t bytes) { size_t r = (off + 255) & ~(size_t)255; off = r + bytes; return r; };
    size_t o_X    = alloc((size_t)MROWS * KP_EMB * 2);
    size_t o_gi   = alloc((size_t)MROWS * G3 * 4);
    size_t o_y0b  = alloc((size_t)MROWS * KP_HID * 2);
    size_t o_gh   = alloc((size_t)N_TWEETS * G3 * 4);
    size_t o_hf0  = alloc((size_t)N_TWEETS * HID * 4);
    size_t o_hf1  = alloc((size_t)N_TWEETS * HID * 4);
    size_t o_h0b0 = alloc((size_t)N_TWEETS * KP_HID * 2);
    size_t o_hbA  = alloc((size_t)N_TWEETS * KP_HID * 2);
    size_t o_hbB  = alloc((size_t)N_TWEETS * KP_HID * 2);
    size_t o_wih0 = alloc((size_t)NP_G3 * KP_EMB * 2);
    size_t o_whh0 = alloc((size_t)NP_G3 * KP_HID * 2);
    size_t o_wih1 = alloc((size_t)NP_G3 * KP_HID * 2);
    size_t o_whh1 = alloc((size_t)NP_G3 * KP_HID * 2);
    size_t o_c1p  = alloc((size_t)NP_H1 * KP_HID * 2);
    size_t o_cmp  = alloc((size_t)NP_H2 * H1C * 2);
    size_t o_clp  = alloc((size_t)NP_H2 * H1C * 2);
    size_t o_ue   = alloc((size_t)N_USERS * HID * 4);
    size_t o_xinb = alloc((size_t)NN * KP_HID * 2);
    size_t o_xw1  = alloc((size_t)NN * H1C * 4);
    size_t o_agg1 = alloc((size_t)NN * H1C * 4);
    size_t o_hfb  = alloc((size_t)NN * H1C * 2);
    size_t o_xwm  = alloc((size_t)NN * H2C * 4);
    size_t o_aggm = alloc((size_t)NN * H2C * 4);
    size_t o_xwl  = alloc((size_t)NN * H2C * 4);
    size_t o_aggl = alloc((size_t)NN * H2C * 4);
    size_t o_mu   = alloc((size_t)NN * H2C * 4);
    size_t o_lv   = alloc((size_t)NN * H2C * 4);
    size_t o_Zb   = alloc((size_t)NN * KP_HID * 2);
    size_t o_deg  = alloc((size_t)NN * 4);
    size_t o_dis  = alloc((size_t)NN * 4);
    size_t o_Ag   = alloc((size_t)NN * NN * 4);
    size_t o_acc  = alloc(8 * 4);
    #define WF(o)  ((float*)(ws + (o)))
    #define WB(o)  ((bf16*)(ws + (o)))

    // -------- zero init ----------
    hipMemsetAsync(ws + o_y0b,  0, (size_t)MROWS * KP_HID * 2, stream);
    hipMemsetAsync(ws + o_hbB,  0, (size_t)N_TWEETS * KP_HID * 2, stream);
    hipMemsetAsync(ws + o_agg1, 0, (size_t)NN * H1C * 4, stream);
    hipMemsetAsync(ws + o_aggm, 0, (size_t)NN * H2C * 4, stream);
    hipMemsetAsync(ws + o_aggl, 0, (size_t)NN * H2C * 4, stream);
    hipMemsetAsync(ws + o_deg,  0, (size_t)NN * 4, stream);
    hipMemsetAsync(ws + o_Ag,   0, (size_t)NN * NN * 4, stream);
    hipMemsetAsync(ws + o_acc,  0, 8 * 4, stream);

    // -------- weight packing ----------
    auto packg = [&](const float* W, int r, int c, int Np, int Kp, int tr, size_t o) {
        int n = Np * Kp;
        k_pack<<<(n + 255) / 256, 256, 0, stream>>>(W, r, c, Np, Kp, tr, WB(o));
    };
    packg(wih0, G3, EMBED, NP_G3, KP_EMB, 0, o_wih0);
    packg(whh0, G3, HID,   NP_G3, KP_HID, 0, o_whh0);
    packg(wih1, G3, HID,   NP_G3, KP_HID, 0, o_wih1);
    packg(whh1, G3, HID,   NP_G3, KP_HID, 0, o_whh1);
    packg(c1w,  HID, H1C,  NP_H1, KP_HID, 1, o_c1p);
    packg(cmw,  H1C, H2C,  NP_H2, H1C,    1, o_cmp);
    packg(clw,  H1C, H2C,  NP_H2, H1C,    1, o_clp);

    // -------- user MLP + embedding gather + h0 conversion ----------
    k_user_mlp<<<N_USERS, 128, 0, stream>>>(uf, w1, b1, w2, b2, WF(o_ue));
    k_embed_gather<<<MROWS, 64, 0, stream>>>(gnf, emb, WB(o_X));
    k_cvt_pad<<<(N_TWEETS * KP_HID + 255) / 256, 256, 0, stream>>>(h0, WB(o_h0b0), N_TWEETS, HID, KP_HID);
    k_cvt_pad<<<(N_TWEETS * KP_HID + 255) / 256, 256, 0, stream>>>(h0 + (size_t)N_TWEETS * HID, WB(o_hbA), N_TWEETS, HID, KP_HID);

    // -------- GRU layer 0 --------
    k_gemm_bf16<<<dim3(MROWS / 64, (G3 + 63) / 64), 256, 0, stream>>>(
        WB(o_X), KP_EMB, WB(o_wih0), KP_EMB, bih0, WF(o_gi), G3, KP_EMB);
    for (int s = 0; s < SEQ; ++s) {
        const bf16* hprevb = (s == 0) ? WB(o_h0b0) : WB(o_y0b) + (size_t)(s - 1) * N_TWEETS * KP_HID;
        k_gemm_bf16<<<dim3(N_TWEETS / 64, (G3 + 63) / 64), 256, 0, stream>>>(
            hprevb, KP_HID, WB(o_whh0), KP_HID, bhh0, WF(o_gh), G3, KP_HID);
        const float* hprev = (s == 0) ? h0 : WF(o_hf0);
        k_gates<<<(N_TWEETS * HID + 255) / 256, 256, 0, stream>>>(
            WF(o_gi) + (size_t)s * N_TWEETS * G3, WF(o_gh), hprev, WF(o_hf0),
            WB(o_y0b) + (size_t)s * N_TWEETS * KP_HID);
    }

    // -------- GRU layer 1 --------
    k_gemm_bf16<<<dim3(MROWS / 64, (G3 + 63) / 64), 256, 0, stream>>>(
        WB(o_y0b), KP_HID, WB(o_wih1), KP_HID, bih1, WF(o_gi), G3, KP_HID);
    for (int s = 0; s < SEQ; ++s) {
        const bf16* hprevb = (s & 1) ? WB(o_hbB) : WB(o_hbA);
        bf16* houtb        = (s & 1) ? WB(o_hbA) : WB(o_hbB);
        k_gemm_bf16<<<dim3(N_TWEETS / 64, (G3 + 63) / 64), 256, 0, stream>>>(
            hprevb, KP_HID, WB(o_whh1), KP_HID, bhh1, WF(o_gh), G3, KP_HID);
        const float* hprev = (s == 0) ? (h0 + (size_t)N_TWEETS * HID) : WF(o_hf1);
        k_gates<<<(N_TWEETS * HID + 255) / 256, 256, 0, stream>>>(
            WF(o_gi) + (size_t)s * N_TWEETS * G3, WF(o_gh), hprev, WF(o_hf1), houtb);
    }

    // -------- graph prep ----------
    k_build_x<<<(NN * KP_HID + 255) / 256, 256, 0, stream>>>(WF(o_hf1), WF(o_ue), WB(o_xinb));
    k_deg<<<(NE + 255) / 256, 256, 0, stream>>>(ei, WF(o_deg));
    k_ground<<<(NE + 255) / 256, 256, 0, stream>>>(ei, WF(o_Ag));
    k_dis<<<(NN + 255) / 256, 256, 0, stream>>>(WF(o_deg), WF(o_dis));

    // -------- GCN conv1 (ELU) ----------
    k_gemm_bf16<<<dim3(NN / 64, 1), 256, 0, stream>>>(
        WB(o_xinb), KP_HID, WB(o_c1p), KP_HID, nullptr, WF(o_xw1), H1C, KP_HID);
    k_scatter<<<(int)(((long)NE * H1C + 255) / 256), 256, 0, stream>>>(
        WF(o_xw1), WF(o_agg1), ei, WF(o_dis), H1C);
    k_final_elu<<<(NN * H1C + 255) / 256, 256, 0, stream>>>(
        WF(o_xw1), WF(o_agg1), WF(o_dis), c1b, WB(o_hfb));

    // -------- GCN conv mu / logvar (ReLU) ----------
    k_gemm_bf16<<<dim3(NN / 64, (H2C + 63) / 64), 256, 0, stream>>>(
        WB(o_hfb), H1C, WB(o_cmp), H1C, nullptr, WF(o_xwm), H2C, H1C);
    k_gemm_bf16<<<dim3(NN / 64, (H2C + 63) / 64), 256, 0, stream>>>(
        WB(o_hfb), H1C, WB(o_clp), H1C, nullptr, WF(o_xwl), H2C, H1C);
    k_scatter<<<(int)(((long)NE * H2C + 255) / 256), 256, 0, stream>>>(
        WF(o_xwm), WF(o_aggm), ei, WF(o_dis), H2C);
    k_scatter<<<(int)(((long)NE * H2C + 255) / 256), 256, 0, stream>>>(
        WF(o_xwl), WF(o_aggl), ei, WF(o_dis), H2C);
    k_final_relu<<<(NN * H2C + 255) / 256, 256, 0, stream>>>(
        WF(o_xwm), WF(o_aggm), WF(o_dis), cmb, WF(o_mu));
    k_final_relu<<<(NN * H2C + 255) / 256, 256, 0, stream>>>(
        WF(o_xwl), WF(o_aggl), WF(o_dis), clb, WF(o_lv));

    // -------- reparametrize + KL + fused ZZ^T loss ----------
    k_z<<<(NN * KP_HID + 255) / 256, 256, 0, stream>>>(
        WF(o_mu), WF(o_lv), eps, WB(o_Zb), out, WF(o_acc));
    k_adj_loss<<<dim3(NN / 64, NN / 64), 256, 0, stream>>>(WB(o_Zb), WF(o_Ag), WF(o_acc));
    k_final<<<1, 32, 0, stream>>>(WF(o_acc), out);
}
